// BitLinearLayer_77558519431327
// MI455X (gfx1250) — 8-bit matrix operand order repaired and checked by reading, compiled, not run on hardware
//
#include <hip/hip_runtime.h>
#include <hip/hip_bf16.h>


typedef __attribute__((ext_vector_type(16))) int          v16i;
typedef __attribute__((ext_vector_type(8)))  float        v8f;
typedef __attribute__((ext_vector_type(4)))  unsigned int u32x4;

#define BATCH 8192
#define KBITS 4096
#define OUTF  4096
#define PKDW  512
#define KT    (KBITS/128)

__device__ __forceinline__ unsigned pm1_fp8(unsigned nib) {
    return ((nib * 0x10204080u) & 0x80808080u) ^ 0xB8B8B8B8u;
}

__device__ __forceinline__ constexpr unsigned KSA(int v) {
    return ((v >> 3) & 1) * 64 + ((v >> 2) & 1) * 32 + ((v >> 1) & 1) * 8 + (v & 1) * 4;
}
__device__ __forceinline__ constexpr unsigned KSB(int v) {
    return (v >> 2) * 32 + (v & 3) * 4;
}

__device__ __forceinline__ void load16(const unsigned* __restrict__ p, unsigned w[16]) {
#pragma unroll
    for (int c = 0; c < 4; ++c) {
        u32x4 q = *(const u32x4*)(p + c * 4);
        w[4 * c + 0] = q[0]; w[4 * c + 1] = q[1];
        w[4 * c + 2] = q[2]; w[4 * c + 3] = q[3];
    }
}

__device__ __forceinline__ v16i unpackA16(const unsigned w[16], int hi) {
    v16i f;
#pragma unroll
    for (int v = 0; v < 16; ++v) {
        const unsigned K0 = KSA(v);
        unsigned word = hi ? w[(K0 >> 3) + 2] : w[K0 >> 3];
        f[v] = (int)pm1_fp8((word >> (K0 & 7)) & 0xFu);
    }
    return f;
}
__device__ __forceinline__ v16i unpackB16(const unsigned w[16], int hi) {
    v16i f;
#pragma unroll
    for (int v = 0; v < 16; ++v) {
        const unsigned K0 = KSB(v);
        unsigned word = hi ? w[(K0 >> 3) + 2] : w[K0 >> 3];
        f[v] = (int)pm1_fp8((word >> (K0 & 7)) & 0xFu);
    }
    return f;
}

__global__ __launch_bounds__(256)
void unpack_fp8_A(const unsigned* __restrict__ pk, unsigned* __restrict__ tiles) {
    const unsigned t    = blockIdx.x * 256u + threadIdx.x;
    const unsigned lane = t & 31u;
    const unsigned tile = t >> 5;
    const unsigned kb   = tile & (KT - 1);
    const unsigned mt   = tile >> 5;
    const unsigned row  = mt * 16 + (lane & 15);
    const int      hi   = (int)(lane >> 4);
    unsigned w[16];
    load16(pk + (size_t)row * PKDW + kb * 16, w);
    v16i f = unpackA16(w, hi);
    unsigned* dst = tiles + (size_t)tile * 512;
#pragma unroll
    for (int c = 0; c < 4; ++c) {
        u32x4 q = {(unsigned)f[4 * c + 0], (unsigned)f[4 * c + 1],
                   (unsigned)f[4 * c + 2], (unsigned)f[4 * c + 3]};
        *(u32x4*)(dst + (c * 32 + lane) * 4) = q;
    }
}

__global__ __launch_bounds__(256)
void unpack_fp8_B(const unsigned* __restrict__ pk, unsigned* __restrict__ tiles) {
    const unsigned t    = blockIdx.x * 256u + threadIdx.x;
    const unsigned lane = t & 31u;
    const unsigned tile = t >> 5;
    const unsigned kb   = tile & (KT - 1);
    const unsigned nt   = tile >> 5;
    const unsigned row  = nt * 16 + (lane & 15);
    const int      hi   = (int)(lane >> 4);
    unsigned w[16];
    load16(pk + (size_t)row * PKDW + kb * 16, w);
    v16i f = unpackB16(w, hi);
    unsigned* dst = tiles + (size_t)tile * 512;
#pragma unroll
    for (int c = 0; c < 4; ++c) {
        u32x4 q = {(unsigned)f[4 * c + 0], (unsigned)f[4 * c + 1],
                   (unsigned)f[4 * c + 2], (unsigned)f[4 * c + 3]};
        *(u32x4*)(dst + (c * 32 + lane) * 4) = q;
    }
}

__global__ __launch_bounds__(256)
void gemm_fp8(const unsigned* __restrict__ At, const unsigned* __restrict__ Bt,
              const float* __restrict__ bias, float* __restrict__ out) {
    const int lane = threadIdx.x & 31;
    const int wave = threadIdx.x >> 5;
    const int wm = wave >> 2;
    const int wn = wave & 3;
    const int lm = lane & 15;
    const int hi = lane >> 4;

    const int mt0 = blockIdx.y * 8 + wm * 4;
    const int nt0 = blockIdx.x * 16 + wn * 4;

    v8f acc[4][4] = {};

    for (int kb = 0; kb < KT; ++kb) {
        v16i a[4];
#pragma unroll
        for (int i = 0; i < 4; ++i) {
            const unsigned* tp = At + ((size_t)(mt0 + i) * KT + kb) * 512;
            __builtin_prefetch(tp + 512, 0, 3);
#pragma unroll
            for (int c = 0; c < 4; ++c) {
                u32x4 q = *(const u32x4*)(tp + (c * 32 + lane) * 4);
                a[i][4 * c + 0] = (int)q[0]; a[i][4 * c + 1] = (int)q[1];
                a[i][4 * c + 2] = (int)q[2]; a[i][4 * c + 3] = (int)q[3];
            }
        }
#pragma unroll
        for (int j = 0; j < 4; ++j) {
            const unsigned* tp = Bt + ((size_t)(nt0 + j) * KT + kb) * 512;
            __builtin_prefetch(tp + 512, 0, 3);
            v16i b;
#pragma unroll
            for (int c = 0; c < 4; ++c) {
                u32x4 q = *(const u32x4*)(tp + (c * 32 + lane) * 4);
                b[4 * c + 0] = (int)q[0]; b[4 * c + 1] = (int)q[1];
                b[4 * c + 2] = (int)q[2]; b[4 * c + 3] = (int)q[3];
            }
#pragma unroll
            for (int i = 0; i < 4; ++i)
                acc[i][j] = __builtin_amdgcn_wmma_f32_16x16x128_fp8_fp8(
                    a[i], b, (short)0, acc[i][j], false, false);
        }
    }

#pragma unroll
    for (int j = 0; j < 4; ++j) {
        const int   col = (nt0 + j) * 16 + lm;
        const float bv  = bias[col];
#pragma unroll
        for (int i = 0; i < 4; ++i) {
            const int row = (mt0 + i) * 16 + hi * 8;
#pragma unroll
            for (int r = 0; r < 8; ++r)
                out[(size_t)(row + r) * OUTF + col] = acc[i][j][r] + bv;
        }
    }
}

__global__ __launch_bounds__(256)
void bitlinear_fused(const unsigned* __restrict__ Apk, const unsigned* __restrict__ Wpk,
                     const float* __restrict__ bias, float* __restrict__ out) {
    const int lane = threadIdx.x & 31;
    const int wave = threadIdx.x >> 5;
    const int wm = wave >> 2;
    const int wn = wave & 3;
    const int lm = lane & 15;
    const int hi = lane >> 4;

    const int m0 = blockIdx.y * 64 + wm * 32;
    const int n0 = blockIdx.x * 128 + wn * 32;

    const unsigned* ar0 = Apk + (size_t)(m0 + lm) * PKDW;
    const unsigned* ar1 = ar0 + 16 * PKDW;
    const unsigned* br0 = Wpk + (size_t)(n0 + lm) * PKDW;
    const unsigned* br1 = br0 + 16 * PKDW;

    v8f acc[2][2] = {};

    for (int kb = 0; kb < KT; ++kb) {
        unsigned wa0[16], wa1[16], wb0[16], wb1[16];
        load16(ar0 + kb * 16, wa0);
        load16(ar1 + kb * 16, wa1);
        load16(br0 + kb * 16, wb0);
        load16(br1 + kb * 16, wb1);
        v16i a0 = unpackA16(wa0, hi), a1 = unpackA16(wa1, hi);
        v16i b0 = unpackB16(wb0, hi), b1 = unpackB16(wb1, hi);
        acc[0][0] = __builtin_amdgcn_wmma_f32_16x16x128_fp8_fp8(a0, b0, (short)0, acc[0][0], false, false);
        acc[0][1] = __builtin_amdgcn_wmma_f32_16x16x128_fp8_fp8(a0, b1, (short)0, acc[0][1], false, false);
        acc[1][0] = __builtin_amdgcn_wmma_f32_16x16x128_fp8_fp8(a1, b0, (short)0, acc[1][0], false, false);
        acc[1][1] = __builtin_amdgcn_wmma_f32_16x16x128_fp8_fp8(a1, b1, (short)0, acc[1][1], false, false);
    }

#pragma unroll
    for (int j = 0; j < 2; ++j) {
        const int   col = n0 + j * 16 + lm;
        const float bv  = bias[col];
#pragma unroll
        for (int i = 0; i < 2; ++i) {
            const int row = m0 + i * 16 + hi * 8;
#pragma unroll
            for (int r = 0; r < 8; ++r)
                out[(size_t)(row + r) * OUTF + col] = acc[i][j][r] + bv;
        }
    }
}

extern "C" void kernel_launch(void* const* d_in, const int* in_sizes, int n_in,
                              void* d_out, int out_size, void* d_ws, size_t ws_size,
                              hipStream_t stream) {
    const unsigned* Apk  = (const unsigned*)d_in[0];
    const unsigned* Wpk  = (const unsigned*)d_in[1];
    const float*    bias = (const float*)d_in[2];
    float*          out  = (float*)d_out;

    const size_t aBytes = (size_t)(BATCH / 16) * KT * 2048;
    const size_t bBytes = (size_t)(OUTF / 16) * KT * 2048;

    if (ws_size >= aBytes + bBytes) {
        unsigned* At = (unsigned*)d_ws;
        unsigned* Bt = (unsigned*)((char*)d_ws + aBytes);
        unpack_fp8_A<<<(BATCH / 16) * KT * 32 / 256, 256, 0, stream>>>(Apk, At);
        unpack_fp8_B<<<(OUTF / 16) * KT * 32 / 256, 256, 0, stream>>>(Wpk, Bt);
        dim3 g(OUTF / 256, BATCH / 128);
        gemm_fp8<<<g, 256, 0, stream>>>(At, Bt, bias, out);
    } else {
        dim3 g(OUTF / 128, BATCH / 64);
        bitlinear_fused<<<g, 256, 0, stream>>>(Apk, Wpk, bias, out);
    }
}
